// MultiheadAttention_77335181132155
// MI455X (gfx1250) — matrix operand order repaired and checked by reading, not compiled, not run on hardware
//
#include <hip/hip_runtime.h>

typedef __bf16 bf16;
typedef __attribute__((ext_vector_type(16))) __bf16 v16bf;
typedef __attribute__((ext_vector_type(8)))  __bf16 v8bf;
typedef __attribute__((ext_vector_type(8)))  float  v8f;

#define DD   1024
#define SS   2048
#define HH   16
#define DH   64
#define BB   2
#define MM   (BB * SS)   // 4096 rows

// ---------- helpers ----------

static __device__ __forceinline__ bf16 f2bf(float f) {
    union { float f; unsigned u; } x;
    x.f = f;
    unsigned r = x.u + 0x7FFFu + ((x.u >> 16) & 1u);   // round-to-nearest-even
    unsigned short h = (unsigned short)(r >> 16);
    return __builtin_bit_cast(bf16, h);
}

// A-matrix fragment (16x32 bf16): lane holds row m=lane%16.
// element e -> K = e + (lane>>4)*16  ==> two contiguous 8-elem chunks:
//   lo @ k0 + half*16, hi @ k0 + 8 + half*16
static __device__ __forceinline__ v16bf load_a_frag(const bf16* lo, const bf16* hi) {
    v8bf l = *(const v8bf*)lo;
    v8bf h = *(const v8bf*)hi;
    v16bf a;
#pragma unroll
    for (int i = 0; i < 8; ++i) { a[i] = l[i]; a[i + 8] = h[i]; }
    return a;
}

static __device__ __forceinline__ v8f wmma_bf16(v16bf a, v16bf b, v8f c) {
    return __builtin_amdgcn_wmma_f32_16x16x32_bf16(
        false, a, false, b, (short)0, c, false, false);
}

// ---------- fp32 -> bf16 conversion ----------

__global__ __launch_bounds__(256) void cvt_bf16(const float* __restrict__ src,
                                                bf16* __restrict__ dst, int n) {
    int i = blockIdx.x * 256 + threadIdx.x;
    if (i < n) dst[i] = f2bf(src[i]);
}

// ---------- GEMM: Y[m,n] = sum_k X[m,k] * W[n,k] + bias[n] ----------
// Per-wave tile: 32 (M) x 64 (N): 2 A-frags + 4 B-frags -> 8 WMMAs per K=32 step.
// mode 0: store bf16 row-major [M, D]
// mode 1: store bf16 V-transposed: [(b*H+h)*64+nd][S]  (n = h*64+nd, m = b*S+s)
// mode 2: store fp32 row-major [M, D]
__global__ __launch_bounds__(128) void gemm_bf16(const bf16* __restrict__ X,
                                                 const bf16* __restrict__ W,
                                                 const float* __restrict__ bias,
                                                 void* __restrict__ Yv, int mode) {
    const int lane = threadIdx.x & 31;
    const int wid  = threadIdx.x >> 5;
    const int tile = blockIdx.x * 4 + wid;          // 2048 tiles total
    const int mt   = tile & 127;                    // 128 M-tiles of 32 rows
    const int ntb  = tile >> 7;                     // 16 N-super-tiles of 64 cols
    const int m0   = mt * 32;
    const int n0   = ntb * 64;
    const int half = lane >> 4;
    const int nl   = lane & 15;

    const bf16* Xrow0 = X + (size_t)(m0 + nl) * DD;        // rows m0..m0+15
    const bf16* Xrow1 = Xrow0 + (size_t)16 * DD;           // rows m0+16..m0+31
    const bf16* Wrow[4];
#pragma unroll
    for (int t = 0; t < 4; ++t)
        Wrow[t] = W + (size_t)(n0 + t * 16 + nl) * DD;

    v8f acc[2][4];
#pragma unroll
    for (int mi = 0; mi < 2; ++mi)
#pragma unroll
        for (int t = 0; t < 4; ++t) acc[mi][t] = (v8f){};

    for (int k0 = 0; k0 < DD; k0 += 32) {
        v16bf a0 = load_a_frag(Xrow0 + k0 + half * 16, Xrow0 + k0 + 8 + half * 16);
        v16bf a1 = load_a_frag(Xrow1 + k0 + half * 16, Xrow1 + k0 + 8 + half * 16);
#pragma unroll
        for (int t = 0; t < 4; ++t) {
            v16bf bt = *(const v16bf*)(Wrow[t] + k0 + half * 16);
            acc[0][t] = wmma_bf16(a0, bt, acc[0][t]);
            acc[1][t] = wmma_bf16(a1, bt, acc[1][t]);
        }
    }

    // bias + store
#pragma unroll
    for (int mi = 0; mi < 2; ++mi) {
#pragma unroll
        for (int t = 0; t < 4; ++t) {
            const int n = n0 + t * 16 + nl;
            const float bv = bias[n];
#pragma unroll
            for (int r = 0; r < 8; ++r) {
                float val = acc[mi][t][r] + bv;
                const int m = m0 + mi * 16 + r + 8 * half;  // C layout: M = r + 8*half
                if (mode == 2) {
                    ((float*)Yv)[(size_t)m * DD + n] = val;
                } else if (mode == 0) {
                    ((bf16*)Yv)[(size_t)m * DD + n] = f2bf(val);
                } else {
                    const int b = m >> 11, s = m & (SS - 1);
                    const int h = n >> 6,  nd = n & 63;
                    ((bf16*)Yv)[(size_t)((b * HH + h) * DH + nd) * SS + s] = f2bf(val);
                }
            }
        }
    }
}

// ---------- flash attention: one wave per (b, h, 16-query tile) ----------
__global__ __launch_bounds__(128) void attn_flash(const bf16* __restrict__ Q,
                                                  const bf16* __restrict__ Km,
                                                  const bf16* __restrict__ Vt,
                                                  bf16* __restrict__ Ctx) {
    __shared__ __align__(16) bf16 Plds[4][16][32];   // 4 KB, one 16x32 P tile per wave

    const int lane = threadIdx.x & 31;
    const int wid  = threadIdx.x >> 5;
    const int id   = blockIdx.x * 4 + wid;           // 4096 waves
    const int qt   = id & 127;
    const int h    = (id >> 7) & (HH - 1);
    const int b    = id >> 11;
    const int half = lane >> 4;
    const int nl   = lane & 15;

    // resident Q fragments (16 queries x 64 dims -> two K=32 A-frags)
    const bf16* Qrow = Q + (size_t)(b * SS + qt * 16 + nl) * DD + h * DH;
    v16bf aq0 = load_a_frag(Qrow + half * 16,      Qrow + 8 + half * 16);
    v16bf aq1 = load_a_frag(Qrow + 32 + half * 16, Qrow + 40 + half * 16);

    float mrow[8], lrow[8];
#pragma unroll
    for (int r = 0; r < 8; ++r) { mrow[r] = -INFINITY; lrow[r] = 0.0f; }
    v8f o0 = {}, o1 = {}, o2 = {}, o3 = {};
    const float scale = 0.125f;                      // 1/sqrt(64)

    for (int kb = 0; kb < SS; kb += 32) {
        // ---- scores for 32 keys (two 16-key column tiles) ----
        const bf16* Kr0 = Km + (size_t)(b * SS + kb + nl) * DD + h * DH;
        const bf16* Kr1 = Kr0 + (size_t)16 * DD;
        v8f s0 = {}, s1 = {};
        s0 = wmma_bf16(aq0, *(const v16bf*)(Kr0 + half * 16),      s0);
        s0 = wmma_bf16(aq1, *(const v16bf*)(Kr0 + 32 + half * 16), s0);
        s1 = wmma_bf16(aq0, *(const v16bf*)(Kr1 + half * 16),      s1);
        s1 = wmma_bf16(aq1, *(const v16bf*)(Kr1 + 32 + half * 16), s1);

        // ---- online softmax (rows live in 16-lane halves: M = r + 8*half) ----
        float p0[8], p1[8];
#pragma unroll
        for (int r = 0; r < 8; ++r) {
            float v0 = s0[r] * scale;
            float v1 = s1[r] * scale;
            float mx = fmaxf(v0, v1);
            mx = fmaxf(mx, __shfl_xor(mx, 1, 32));
            mx = fmaxf(mx, __shfl_xor(mx, 2, 32));
            mx = fmaxf(mx, __shfl_xor(mx, 4, 32));
            mx = fmaxf(mx, __shfl_xor(mx, 8, 32));
            const float mnew  = fmaxf(mrow[r], mx);
            const float alpha = __expf(mrow[r] - mnew);
            v0 = __expf(v0 - mnew);
            v1 = __expf(v1 - mnew);
            float sum = v0 + v1;
            sum += __shfl_xor(sum, 1, 32);
            sum += __shfl_xor(sum, 2, 32);
            sum += __shfl_xor(sum, 4, 32);
            sum += __shfl_xor(sum, 8, 32);
            lrow[r] = lrow[r] * alpha + sum;
            mrow[r] = mnew;
            p0[r] = v0; p1[r] = v1;
            o0[r] *= alpha; o1[r] *= alpha; o2[r] *= alpha; o3[r] *= alpha;
        }

        // ---- C-layout -> A-layout for P via per-wave LDS tile ----
        __syncthreads();                              // WAR guard vs previous loads
#pragma unroll
        for (int r = 0; r < 8; ++r) {
            Plds[wid][r + 8 * half][nl]      = f2bf(p0[r]);
            Plds[wid][r + 8 * half][16 + nl] = f2bf(p1[r]);
        }
        __syncthreads();                              // make stores visible to loads
        v16bf pa = load_a_frag(&Plds[wid][nl][half * 16],
                               &Plds[wid][nl][8 + half * 16]);

        // ---- P @ V : V stored transposed [b,h,d][S] so B-frag is contiguous ----
        const bf16* Vbase = Vt + (size_t)((b * HH + h) * DH + nl) * SS + kb + half * 16;
        o0 = wmma_bf16(pa, *(const v16bf*)(Vbase + (size_t)0 * 16 * SS), o0);
        o1 = wmma_bf16(pa, *(const v16bf*)(Vbase + (size_t)1 * 16 * SS), o1);
        o2 = wmma_bf16(pa, *(const v16bf*)(Vbase + (size_t)2 * 16 * SS), o2);
        o3 = wmma_bf16(pa, *(const v16bf*)(Vbase + (size_t)3 * 16 * SS), o3);
    }

    // ---- normalize and store context (bf16, [B,S,D] layout) ----
#pragma unroll
    for (int r = 0; r < 8; ++r) {
        const float inv = 1.0f / lrow[r];
        const int s = qt * 16 + r + 8 * half;
        bf16* crow = Ctx + (size_t)(b * SS + s) * DD + h * DH;
        crow[0 * 16 + nl] = f2bf(o0[r] * inv);
        crow[1 * 16 + nl] = f2bf(o1[r] * inv);
        crow[2 * 16 + nl] = f2bf(o2[r] * inv);
        crow[3 * 16 + nl] = f2bf(o3[r] * inv);
    }
}

// ---------- host launcher ----------

extern "C" void kernel_launch(void* const* d_in, const int* in_sizes, int n_in,
                              void* d_out, int out_size, void* d_ws, size_t ws_size,
                              hipStream_t stream) {
    (void)in_sizes; (void)n_in; (void)out_size; (void)ws_size;

    const float* x_q   = (const float*)d_in[0];
    const float* x_kv  = (const float*)d_in[1];
    const float* W_q   = (const float*)d_in[2];
    const float* b_q   = (const float*)d_in[3];
    const float* W_k   = (const float*)d_in[4];
    const float* b_k   = (const float*)d_in[5];
    const float* W_v   = (const float*)d_in[6];
    const float* b_v   = (const float*)d_in[7];
    const float* W_out = (const float*)d_in[8];
    const float* b_out = (const float*)d_in[9];
    float* out = (float*)d_out;

    const size_t WSZ = (size_t)DD * DD;   // 1M elems
    const size_t XSZ = (size_t)MM * DD;   // 4M elems

    bf16* w    = (bf16*)d_ws;
    bf16* Wq   = w;
    bf16* Wk   = w + 1 * WSZ;
    bf16* Wv   = w + 2 * WSZ;
    bf16* Wo   = w + 3 * WSZ;
    bf16* Xq   = w + 4 * WSZ;
    bf16* Xkv  = Xq  + XSZ;
    bf16* Qb   = Xkv + XSZ;
    bf16* Kb   = Qb  + XSZ;
    bf16* Vt   = Kb  + XSZ;
    bf16* Ctx  = Vt  + XSZ;

    // fp32 -> bf16 conversions
    cvt_bf16<<<(int)((WSZ + 255) / 256), 256, 0, stream>>>(W_q,   Wq,  (int)WSZ);
    cvt_bf16<<<(int)((WSZ + 255) / 256), 256, 0, stream>>>(W_k,   Wk,  (int)WSZ);
    cvt_bf16<<<(int)((WSZ + 255) / 256), 256, 0, stream>>>(W_v,   Wv,  (int)WSZ);
    cvt_bf16<<<(int)((WSZ + 255) / 256), 256, 0, stream>>>(W_out, Wo,  (int)WSZ);
    cvt_bf16<<<(int)((XSZ + 255) / 256), 256, 0, stream>>>(x_q,   Xq,  (int)XSZ);
    cvt_bf16<<<(int)((XSZ + 255) / 256), 256, 0, stream>>>(x_kv,  Xkv, (int)XSZ);

    // Q/K/V projections: 128 x 16 = 2048 tiles -> 512 blocks of 4 waves
    gemm_bf16<<<512, 128, 0, stream>>>(Xq,  Wq, b_q, (void*)Qb, 0);
    gemm_bf16<<<512, 128, 0, stream>>>(Xkv, Wk, b_k, (void*)Kb, 0);
    gemm_bf16<<<512, 128, 0, stream>>>(Xkv, Wv, b_v, (void*)Vt, 1);   // transposed store

    // attention: 2*16*128 = 4096 waves -> 1024 blocks of 4 waves
    attn_flash<<<1024, 128, 0, stream>>>(Qb, Kb, Vt, Ctx);

    // output projection -> fp32
    gemm_bf16<<<512, 128, 0, stream>>>(Ctx, Wo, b_out, (void*)out, 2);
}
